// LSTMCELL_81183471829582
// MI455X (gfx1250) — matrix operand order repaired and checked by reading, not compiled, not run on hardware
//
#include <hip/hip_runtime.h>

// Problem dims (from reference)
#define B_DIM 4096
#define D_DIM 1024
#define H_DIM 2048
#define O_DIM 2048

typedef unsigned short ushort_t;
typedef unsigned int   uint_t;
typedef __attribute__((ext_vector_type(16))) __bf16 v16bf;
typedef __attribute__((ext_vector_type(8)))  float  v8f;
typedef __attribute__((ext_vector_type(4)))  unsigned int u32x4;
typedef __attribute__((ext_vector_type(8)))  int i32x8;
typedef __attribute__((ext_vector_type(4)))  int i32x4;

#define LDS_STRIDE   40                 // bf16 elems per LDS row (80 B, 16B aligned)
#define A_TILE_ELEMS (64 * LDS_STRIDE)
#define TILE_W_ELEMS (32 * LDS_STRIDE)

#if __has_builtin(__builtin_amdgcn_tensor_load_to_lds)
#define HAVE_TDM 1
#else
#define HAVE_TDM 0
#endif

// pack two f32 -> packed bf16x2; plain casts let the backend pick a native cvt
__device__ __forceinline__ uint_t pk2(float x, float y) {
  __bf16 lo = (__bf16)x;
  __bf16 hi = (__bf16)y;
  return (uint_t)__builtin_bit_cast(ushort_t, lo) |
         ((uint_t)__builtin_bit_cast(ushort_t, hi) << 16);
}
__device__ __forceinline__ ushort_t f2bf(float x) {
  return __builtin_bit_cast(ushort_t, (__bf16)x);
}

__device__ __forceinline__ v8f wmma_bf16(v16bf a, v16bf b, v8f c) {
  return __builtin_amdgcn_wmma_f32_16x16x32_bf16(
      /*neg_a=*/false, a, /*neg_b=*/false, b,
      /*c_mod=*/(short)0, c, /*reuse_a=*/false, /*reuse_b=*/false);
}

__device__ __forceinline__ void wait_async0() {
#if __has_builtin(__builtin_amdgcn_s_wait_asynccnt)
  __builtin_amdgcn_s_wait_asynccnt(0);
#else
  asm volatile("s_wait_asynccnt 0x0" ::: "memory");
#endif
}
__device__ __forceinline__ void wait_tensor0() {
  __builtin_amdgcn_s_wait_tensorcnt((short)0);
}

// ---- TDM: 64x32 bf16 tile (row stride H) -> LDS with stride-40 padding ---
// D# group0/group1 per CDNA5 ISA 8.3/8.4. pad: every 16 DWORDs add 4 DWORDs
// -> LDS row pitch 80 B == LDS_STRIDE bf16 elems.
__device__ __forceinline__ void tdm_load_tile_a(const ushort_t* gsrc, unsigned lds_off) {
#if HAVE_TDM
  unsigned long long ga = (unsigned long long)(size_t)gsrc;
  u32x4 g0;
  g0[0] = 1u;                                            // count=1 (valid user D#)
  g0[1] = lds_off;                                       // lds_addr (bytes)
  g0[2] = (unsigned)(ga & 0xffffffffu);                  // global_addr[31:0]
  g0[3] = (unsigned)((ga >> 32) & 0x01ffffffu)           // global_addr[56:32]
        | (2u << 30);                                    // type=2 ("image")
  i32x8 g1;
  g1[0] = (int)((1u << 16)                               // data_size = 2 bytes
              | (1u << 20)                               // pad_enable
              | (3u << 22)                               // pad_interval: 16 DWORDs
              | (3u << 25));                             // pad_amount : 4 DWORDs
  g1[1] = (int)(((unsigned)H_DIM & 0xffffu) << 16);      // tensor_dim0 lo16
  g1[2] = (int)((((unsigned)H_DIM >> 16) & 0xffffu)      // tensor_dim0 hi16
              | (((unsigned)B_DIM & 0xffffu) << 16));    // tensor_dim1 lo16
  g1[3] = (int)((((unsigned)B_DIM >> 16) & 0xffffu)      // tensor_dim1 hi16
              | (32u << 16));                            // tile_dim0 = 32
  g1[4] = (int)64u;                                      // tile_dim1 = 64, tile_dim2 = 0
  g1[5] = (int)H_DIM;                                    // tensor_dim0_stride lo32
  g1[6] = 0;
  g1[7] = 0;
  i32x4 z4 = {0, 0, 0, 0};
  i32x8 z8 = {0, 0, 0, 0, 0, 0, 0, 0};
  __builtin_amdgcn_tensor_load_to_lds(g0, g1, z4, z4, z8, 0);  // 6-arg form
#else
  (void)gsrc; (void)lds_off;
#endif
}

// ---- register-staged tiles (fp32 global -> packed bf16 regs -> LDS) ------

struct ARegs  { uint4 q; };
struct W4Regs { uint4 q0, q1; };
struct W1Regs { uint2 q; };

__device__ __forceinline__ ARegs load_a(const float* __restrict__ A, int ld,
                                        int m0, int k0, int tid) {
  const int r  = tid >> 2;
  const int c0 = (tid & 3) * 8;
  const float* src = A + (size_t)(m0 + r) * ld + k0 + c0;
  __builtin_prefetch(src + 64, 0, 0);
  float4 v0 = *(const float4*)(src);
  float4 v1 = *(const float4*)(src + 4);
  ARegs a;
  a.q.x = pk2(v0.x, v0.y); a.q.y = pk2(v0.z, v0.w);
  a.q.z = pk2(v1.x, v1.y); a.q.w = pk2(v1.z, v1.w);
  return a;
}
__device__ __forceinline__ void store_a(ushort_t* sA, int tid, ARegs a) {
  const int r  = tid >> 2;
  const int c0 = (tid & 3) * 8;
  *(uint4*)&sA[r * LDS_STRIDE + c0] = a.q;
}

__device__ __forceinline__ W4Regs load_w4(const float* __restrict__ W, int ld,
                                          int n0, int k0, int tid) {
  const int g  = tid >> 6;
  const int r  = (tid >> 1) & 31;
  const int c0 = (tid & 1) * 16;
  const float* src = W + (size_t)(g * H_DIM + n0 + r) * ld + k0 + c0;
  __builtin_prefetch(src + 64, 0, 0);
  float4 v0 = *(const float4*)(src + 0);
  float4 v1 = *(const float4*)(src + 4);
  float4 v2 = *(const float4*)(src + 8);
  float4 v3 = *(const float4*)(src + 12);
  W4Regs w;
  w.q0.x = pk2(v0.x, v0.y); w.q0.y = pk2(v0.z, v0.w);
  w.q0.z = pk2(v1.x, v1.y); w.q0.w = pk2(v1.z, v1.w);
  w.q1.x = pk2(v2.x, v2.y); w.q1.y = pk2(v2.z, v2.w);
  w.q1.z = pk2(v3.x, v3.y); w.q1.w = pk2(v3.z, v3.w);
  return w;
}
__device__ __forceinline__ void store_w4(ushort_t* sW, int tid, W4Regs w) {
  const int g  = tid >> 6;
  const int r  = (tid >> 1) & 31;
  const int c0 = (tid & 1) * 16;
  ushort_t* d = &sW[g * TILE_W_ELEMS + r * LDS_STRIDE + c0];
  *(uint4*)(d)     = w.q0;
  *(uint4*)(d + 8) = w.q1;
}

__device__ __forceinline__ W1Regs load_w1(const float* __restrict__ W, int ld,
                                          int n0, int k0, int tid) {
  const int r  = tid >> 3;
  const int c0 = (tid & 7) * 4;
  const float* src = W + (size_t)(n0 + r) * ld + k0 + c0;
  __builtin_prefetch(src + 64, 0, 0);
  float4 v = *(const float4*)src;
  W1Regs w;
  w.q.x = pk2(v.x, v.y); w.q.y = pk2(v.z, v.w);
  return w;
}
__device__ __forceinline__ void store_w1(ushort_t* sW, int tid, W1Regs w) {
  const int r  = tid >> 3;
  const int c0 = (tid & 7) * 4;
  *(uint2*)&sW[r * LDS_STRIDE + c0] = w.q;
}

// per-lane async fallback: bf16 A tile global -> LDS, 16B per lane
__device__ __forceinline__ void async_copy_a(const ushort_t* __restrict__ hsrc,
                                             int m0, int k0, ushort_t* sA, int tid) {
  const int r  = tid >> 2;
  const int c0 = (tid & 3) * 8;
  const ushort_t* src = hsrc + (size_t)(m0 + r) * H_DIM + k0 + c0;
  unsigned lds = (unsigned)(size_t)(sA + r * LDS_STRIDE + c0);
  unsigned long long g = (unsigned long long)(size_t)src;
  asm volatile("global_load_async_to_lds_b128 %0, %1, off"
               :: "v"(lds), "v"(g) : "memory");
}

// ---- WMMA fragment loads (CDNA5 ISA 7.12.2 layouts) ----------------------

__device__ __forceinline__ v16bf frag_a(const ushort_t* sA, int m_sub, int lane) {
  const int row = m_sub * 16 + (lane & 15);
  const int kb  = (lane >> 4) * 8;
  const ushort_t* p = &sA[row * LDS_STRIDE + kb];
  union { v16bf v; uint4 q[2]; } u;
  u.q[0] = *(const uint4*)(p);
  u.q[1] = *(const uint4*)(p + 16);
  return u.v;
}

__device__ __forceinline__ v16bf frag_b(const ushort_t* sW, int n_sub, int lane) {
  const int row = n_sub * 16 + (lane & 15);
  const int kb  = (lane >> 4) * 8;
  const ushort_t* p = &sW[row * LDS_STRIDE + kb];
  union { v16bf v; uint4 q[2]; } u;
  u.q[0] = *(const uint4*)(p);
  u.q[1] = *(const uint4*)(p + 16);
  return u.v;
}

__device__ __forceinline__ float fast_sigmoid(float x) { return 1.0f / (1.0f + __expf(-x)); }
__device__ __forceinline__ float fast_tanh(float x)    { return 1.0f - 2.0f / (1.0f + __expf(2.0f * x)); }

// ---- double-buffered 4-gate GEMM phase -----------------------------------
__device__ __forceinline__ void gate_phase(
    const float* __restrict__ A, const float* __restrict__ W, int K,
    int m0, int n0, ushort_t* sA, ushort_t* sW,
    int tid, int lane, int m_sub, int n_sub,
    v8f& ai, v8f& af, v8f& ag, v8f& ao) {
  __syncthreads();
  ARegs  ar = load_a(A, K, m0, 0, tid);
  W4Regs wr = load_w4(W, K, n0, 0, tid);
  store_a(sA, tid, ar);
  store_w4(sW, tid, wr);
  const int nb = K / 32;
  for (int i = 0; i < nb; ++i) {
    const int cur = i & 1, nxt = cur ^ 1;
    __syncthreads();
    const bool more = (i + 1 < nb);
    if (more) {
      ar = load_a(A, K, m0, (i + 1) * 32, tid);
      wr = load_w4(W, K, n0, (i + 1) * 32, tid);
    }
    ushort_t* cA = sA + cur * A_TILE_ELEMS;
    ushort_t* cW = sW + cur * 4 * TILE_W_ELEMS;
    v16bf a = frag_a(cA, m_sub, lane);
    ai = wmma_bf16(a, frag_b(cW + 0 * TILE_W_ELEMS, n_sub, lane), ai);
    af = wmma_bf16(a, frag_b(cW + 1 * TILE_W_ELEMS, n_sub, lane), af);
    ag = wmma_bf16(a, frag_b(cW + 2 * TILE_W_ELEMS, n_sub, lane), ag);
    ao = wmma_bf16(a, frag_b(cW + 3 * TILE_W_ELEMS, n_sub, lane), ao);
    if (more) {
      store_a(sA + nxt * A_TILE_ELEMS, tid, ar);
      store_w4(sW + nxt * 4 * TILE_W_ELEMS, tid, wr);
    }
  }
}

// ---- Kernel 1: fused gates + feedback GEMMs + LSTM elementwise -----------
__global__ __launch_bounds__(256) void lstm_cell_kernel(
    const float* __restrict__ data,  const float* __restrict__ h_prev,
    const float* __restrict__ c_prev, const float* __restrict__ o_prev,
    const float* __restrict__ Wi,    const float* __restrict__ bi,
    const float* __restrict__ Wh,    const float* __restrict__ W_fb,
    const float* __restrict__ b_fb,
    float* __restrict__ out_h, float* __restrict__ out_c,
    ushort_t* __restrict__ h_bf) {
  __shared__ ushort_t sA[2 * A_TILE_ELEMS];
  __shared__ ushort_t sW[2 * 4 * TILE_W_ELEMS];

  const int tid   = threadIdx.x;
  const int lane  = tid & 31;
  const int wave  = tid >> 5;
  const int m_sub = wave >> 1;
  const int n_sub = wave & 1;
  const int m0 = blockIdx.y * 64;
  const int n0 = blockIdx.x * 32;

  v8f acc_i = {}, acc_f = {}, acc_g = {}, acc_o = {}, acc_fb = {};

  gate_phase(data,   Wi, D_DIM, m0, n0, sA, sW, tid, lane, m_sub, n_sub,
             acc_i, acc_f, acc_g, acc_o);
  gate_phase(h_prev, Wh, H_DIM, m0, n0, sA, sW, tid, lane, m_sub, n_sub,
             acc_i, acc_f, acc_g, acc_o);

  // Phase 3: o_prev @ W_fb.T (single-tile W, double buffered)
  {
    __syncthreads();
    ARegs  ar = load_a(o_prev, H_DIM, m0, 0, tid);
    W1Regs wr = load_w1(W_fb, H_DIM, n0, 0, tid);
    store_a(sA, tid, ar);
    store_w1(sW, tid, wr);
    const int nb = H_DIM / 32;
    for (int i = 0; i < nb; ++i) {
      const int cur = i & 1, nxt = cur ^ 1;
      __syncthreads();
      const bool more = (i + 1 < nb);
      if (more) {
        ar = load_a(o_prev, H_DIM, m0, (i + 1) * 32, tid);
        wr = load_w1(W_fb, H_DIM, n0, (i + 1) * 32, tid);
      }
      acc_fb = wmma_bf16(frag_a(sA + cur * A_TILE_ELEMS, m_sub, lane),
                         frag_b(sW + cur * 4 * TILE_W_ELEMS, n_sub, lane), acc_fb);
      if (more) {
        store_a(sA + nxt * A_TILE_ELEMS, tid, ar);
        store_w1(sW + nxt * 4 * TILE_W_ELEMS, tid, wr);
      }
    }
  }

  // Epilogue: C/D layout -> (m = r + 8*(lane>=16), n = lane&15)
  const int gm   = m0 + m_sub * 16;
  const int gn   = n0 + n_sub * 16;
  const int nloc = lane & 15;
  const int mhi  = (lane >> 4) * 8;
  const int col  = gn + nloc;
  const float b_i = bi[0 * H_DIM + col];
  const float b_f = bi[1 * H_DIM + col];
  const float b_g = bi[2 * H_DIM + col];
  const float b_o = bi[3 * H_DIM + col];
  const float b_c = b_fb[col];
#pragma unroll
  for (int r = 0; r < 8; ++r) {
    const size_t idx = (size_t)(gm + mhi + r) * H_DIM + col;
    float it = fast_sigmoid(acc_i[r] + b_i);
    float ft = fast_sigmoid(acc_f[r] + b_f);
    float gt = fast_tanh(acc_g[r] + b_g);
    float ot = fast_sigmoid(acc_o[r] + b_o);
    float ct = ft * c_prev[idx] + it * gt + acc_fb[r] + b_c;
    float ht = ot * fast_tanh(ct);
    __builtin_nontemporal_store(ct, &out_c[idx]);   // streamed, never re-read
    __builtin_nontemporal_store(ht, &out_h[idx]);
    h_bf[idx] = f2bf(ht);                           // re-read by out_gemm: keep RT
  }
}

// ---- Kernel 2: logits = h_t(bf16) @ W_out.T + b_out ----------------------
// A tile staged by TDM tensor_load_to_lds (TENSORcnt) when available,
// else per-lane GLOBAL_LOAD_ASYNC_TO_LDS (ASYNCcnt).
__global__ __launch_bounds__(256) void out_gemm_kernel(
    const ushort_t* __restrict__ h_bf, const float* __restrict__ W_out,
    const float* __restrict__ b_out, float* __restrict__ logits) {
  __shared__ ushort_t sA[2 * A_TILE_ELEMS];
  __shared__ ushort_t sW[2 * TILE_W_ELEMS];
  const int tid   = threadIdx.x;
  const int lane  = tid & 31;
  const int wave  = tid >> 5;
  const int m_sub = wave >> 1;
  const int n_sub = wave & 1;
  const int m0 = blockIdx.y * 64;
  const int n0 = blockIdx.x * 32;
  v8f acc = {};

  const ushort_t* aRowBase = h_bf + (size_t)m0 * H_DIM;
  const unsigned  ldsABase = (unsigned)(size_t)sA;

#if HAVE_TDM
  if (wave == 0) tdm_load_tile_a(aRowBase + 0, ldsABase);
#else
  async_copy_a(h_bf, m0, 0, sA, tid);
#endif
  W1Regs wr = load_w1(W_out, H_DIM, n0, 0, tid);
  store_w1(sW, tid, wr);
#if HAVE_TDM
  if (wave == 0) wait_tensor0();
#else
  wait_async0();
#endif
  const int nb = H_DIM / 32;
  for (int i = 0; i < nb; ++i) {
    const int cur = i & 1, nxt = cur ^ 1;
    __syncthreads();
    const bool more = (i + 1 < nb);
    if (more) {
#if HAVE_TDM
      if (wave == 0)
        tdm_load_tile_a(aRowBase + (i + 1) * 32,
                        ldsABase + (unsigned)(nxt * A_TILE_ELEMS * 2));
#else
      async_copy_a(h_bf, m0, (i + 1) * 32, sA + nxt * A_TILE_ELEMS, tid);
#endif
      wr = load_w1(W_out, H_DIM, n0, (i + 1) * 32, tid);
    }
    acc = wmma_bf16(frag_a(sA + cur * A_TILE_ELEMS, m_sub, lane),
                    frag_b(sW + cur * TILE_W_ELEMS, n_sub, lane), acc);
    if (more) {
      store_w1(sW + nxt * TILE_W_ELEMS, tid, wr);
#if HAVE_TDM
      if (wave == 0) wait_tensor0();   // DMA into `nxt` complete before publish
#else
      wait_async0();
#endif
    }
  }
  const int gm = m0 + m_sub * 16, gn = n0 + n_sub * 16;
  const int nloc = lane & 15, mhi = (lane >> 4) * 8;
  const float bb = b_out[gn + nloc];
#pragma unroll
  for (int r = 0; r < 8; ++r)
    logits[(size_t)(gm + mhi + r) * O_DIM + gn + nloc] = acc[r] + bb;
}

// ---- Kernel 3: row softmax over O=2048 -----------------------------------
__global__ __launch_bounds__(256) void softmax_kernel(
    const float* __restrict__ logits, float* __restrict__ out) {
  const int row = blockIdx.x;
  const int tid = threadIdx.x;
  const float* lr = logits + (size_t)row * O_DIM;
  float v[8];
  float m = -3.402823466e+38f;
#pragma unroll
  for (int j = 0; j < 8; ++j) { v[j] = lr[tid + 256 * j]; m = fmaxf(m, v[j]); }
#pragma unroll
  for (int s = 16; s > 0; s >>= 1) m = fmaxf(m, __shfl_xor(m, s, 32));
  __shared__ float red[8];
  __shared__ float bc[2];
  const int wv = tid >> 5, ln = tid & 31;
  if (ln == 0) red[wv] = m;
  __syncthreads();
  if (tid == 0) {
    float t = red[0];
    for (int i = 1; i < 8; ++i) t = fmaxf(t, red[i]);
    bc[0] = t;
  }
  __syncthreads();
  const float rmax = bc[0];
  float s = 0.f;
#pragma unroll
  for (int j = 0; j < 8; ++j) { v[j] = __expf(v[j] - rmax); s += v[j]; }
#pragma unroll
  for (int sh = 16; sh > 0; sh >>= 1) s += __shfl_xor(s, sh, 32);
  if (ln == 0) red[wv] = s;
  __syncthreads();
  if (tid == 0) {
    float t = 0.f;
    for (int i = 0; i < 8; ++i) t += red[i];
    bc[1] = t;
  }
  __syncthreads();
  const float inv = 1.0f / bc[1];
  float* orow = out + (size_t)row * O_DIM;
#pragma unroll
  for (int j = 0; j < 8; ++j)
    __builtin_nontemporal_store(v[j] * inv, &orow[tid + 256 * j]);
}

// ---- launch ---------------------------------------------------------------
extern "C" void kernel_launch(void* const* d_in, const int* in_sizes, int n_in,
                              void* d_out, int out_size, void* d_ws, size_t ws_size,
                              hipStream_t stream) {
  (void)in_sizes; (void)n_in; (void)out_size; (void)ws_size;
  const float* data   = (const float*)d_in[0];
  const float* h_prev = (const float*)d_in[1];
  const float* c_prev = (const float*)d_in[2];
  const float* o_prev = (const float*)d_in[3];
  const float* Wi     = (const float*)d_in[4];
  const float* bi     = (const float*)d_in[5];
  const float* Wh     = (const float*)d_in[6];
  const float* W_out  = (const float*)d_in[7];
  const float* b_out  = (const float*)d_in[8];
  const float* W_fb   = (const float*)d_in[9];
  const float* b_fb   = (const float*)d_in[10];

  float* out_sm = (float*)d_out;                              // [B, O]
  float* out_h  = (float*)d_out + (size_t)B_DIM * O_DIM;      // [B, H]
  float* out_c  = out_h + (size_t)B_DIM * H_DIM;              // [B, H]

  // ws: [B*H bf16 h_t copy][B*O f32 logits]  (~50 MB)
  ushort_t* h_bf  = (ushort_t*)d_ws;
  float*    logit = (float*)((char*)d_ws + (size_t)B_DIM * H_DIM * sizeof(ushort_t));

  dim3 blk(256);
  dim3 grid1(H_DIM / 32, B_DIM / 64);
  hipLaunchKernelGGL(lstm_cell_kernel, grid1, blk, 0, stream,
                     data, h_prev, c_prev, o_prev, Wi, bi, Wh, W_fb, b_fb,
                     out_h, out_c, h_bf);

  dim3 grid2(O_DIM / 32, B_DIM / 64);
  hipLaunchKernelGGL(out_gemm_kernel, grid2, blk, 0, stream,
                     h_bf, W_out, b_out, logit);

  dim3 grid3(B_DIM);
  hipLaunchKernelGGL(softmax_kernel, grid3, blk, 0, stream, logit, out_sm);
}
